// SimplifiedSSMBlock_42898133353243
// MI455X (gfx1250) — matrix operand order repaired and checked by reading, not compiled, not run on hardware
//
#include <hip/hip_runtime.h>
#include <hip/hip_bf16.h>
#include <math.h>

// ---------------------------------------------------------------------------
// SimplifiedSSMBlock for MI455X (gfx1250, wave32, WMMA + async LDS copy)
//   B=2, L=2048, D=1024.  M = B*L = 4096 rows for the GEMM.
// Pipeline:
//   1) scan_partial : per-(b,d,chunk) weighted partial sums of A^s * x
//   2) scan_prefix  : exclusive prefix over chunks (tiny)
//   3) scan_final   : finish scan, conv output, + D*x, exact GELU -> h (f16)
//   4) gemm_wmma    : r = x + h @ W^T + b  via v_wmma_f32_16x16x32_f16,
//                     double-buffered; A staged with global_load_async_to_lds_b128
//   5) layernorm    : in-place LN over D on d_out
// ---------------------------------------------------------------------------

typedef __attribute__((ext_vector_type(16))) _Float16 v16h;
typedef __attribute__((ext_vector_type(8)))  _Float16 v8h;
typedef __attribute__((ext_vector_type(4)))  _Float16 v4h;
typedef __attribute__((ext_vector_type(8)))  float    v8f;

namespace cfg {
constexpr int Bsz  = 2;
constexpr int L    = 2048;
constexpr int D    = 1024;
constexpr int M    = Bsz * L;      // 4096 GEMM rows
constexpr int NCH  = 32;           // scan chunks per (b,d)
constexpr int CHL  = L / NCH;      // 64 elements per chunk
// workspace layout (bytes)
constexpr size_t H_OFF    = 0;                         // h as f16: M*D*2 = 8 MB
constexpr size_t H_BYTES  = (size_t)M * D * 2;
constexpr size_t PART_OFF = H_BYTES;                   // B*NCH*D floats
constexpr size_t PART_BYTES = (size_t)Bsz * NCH * D * 4;
constexpr size_t PFX_OFF  = PART_OFF + PART_BYTES;
}

__device__ __forceinline__ float sigmoidf_(float a) { return 1.0f / (1.0f + expf(-a)); }
__device__ __forceinline__ float gelu_exact(float v) {
  return 0.5f * v * (1.0f + erff(v * 0.70710678118654752f));
}

// Generic (flat) pointer -> raw 32-bit LDS byte offset for async-LDS ops.
__device__ __forceinline__ unsigned lds_off32(const void* p) {
  return (unsigned)(uintptr_t)(const __attribute__((address_space(3))) void*)p;
}

// ---------------------------------------------------------------------------
// Kernel 1: per-chunk weighted partials  P[b][c][d] = sum_{l in chunk} A^l x[b,l,d]
// ---------------------------------------------------------------------------
__global__ __launch_bounds__(256) void scan_partial_k(
    const float* __restrict__ x, const float* __restrict__ A_param,
    float* __restrict__ part) {
  using namespace cfg;
  int idx = blockIdx.x * blockDim.x + threadIdx.x;   // B*NCH*D threads
  int d = idx % D;
  int t = idx / D;
  int c = t % NCH;
  int b = t / NCH;
  float A = sigmoidf_(A_param[d]);
  float p = powf(A, (float)(c * CHL));               // A^(chunk start)
  float acc = 0.0f;
  const float* xp = x + ((size_t)b * L + c * CHL) * D + d;
  #pragma unroll 4
  for (int i = 0; i < CHL; ++i) {
    acc = fmaf(p, xp[(size_t)i * D], acc);
    p *= A;
  }
  part[((size_t)b * NCH + c) * D + d] = acc;
}

// ---------------------------------------------------------------------------
// Kernel 2: exclusive prefix over chunks (B*D threads, NCH-long serial loop)
// ---------------------------------------------------------------------------
__global__ __launch_bounds__(256) void scan_prefix_k(
    const float* __restrict__ part, float* __restrict__ pfx) {
  using namespace cfg;
  int idx = blockIdx.x * blockDim.x + threadIdx.x;   // B*D threads
  int d = idx % D;
  int b = idx / D;
  float run = 0.0f;
  for (int c = 0; c < NCH; ++c) {
    size_t o = ((size_t)b * NCH + c) * D + d;        // coalesced in d
    pfx[o] = run;
    run += part[o];
  }
}

// ---------------------------------------------------------------------------
// Kernel 3: finalize scan, apply decay, +D*x, exact GELU, store h (f16)
//   out[l] = C*B*A^(L-1-l) * S[l] + Dv*x[l];  h = gelu(out)
// ---------------------------------------------------------------------------
__global__ __launch_bounds__(256) void scan_final_k(
    const float* __restrict__ x, const float* __restrict__ A_param,
    const float* __restrict__ B_vec, const float* __restrict__ C_vec,
    const float* __restrict__ D_vec, const float* __restrict__ pfx,
    _Float16* __restrict__ h) {
  using namespace cfg;
  int idx = blockIdx.x * blockDim.x + threadIdx.x;   // B*NCH*D threads
  int d = idx % D;
  int t = idx / D;
  int c = t % NCH;
  int b = t / NCH;
  float A = sigmoidf_(A_param[d]);
  float l2A = log2f(A);
  float CB = C_vec[d] * B_vec[d];
  float Dv = D_vec[d];
  float S = pfx[((size_t)b * NCH + c) * D + d];
  float p = powf(A, (float)(c * CHL));
  const float* xp = x + ((size_t)b * L + c * CHL) * D + d;
  _Float16*   hp = h + ((size_t)b * L + c * CHL) * D + d;
  for (int i = 0; i < CHL; ++i) {
    int l = c * CHL + i;
    float xv = xp[(size_t)i * D];
    S = fmaf(p, xv, S);
    p *= A;
    float q = exp2f((float)(L - 1 - l) * l2A);       // A^(L-1-l)
    float o = fmaf(CB * q, S, Dv * xv);
    hp[(size_t)i * D] = (_Float16)gelu_exact(o);
  }
}

// ---------------------------------------------------------------------------
// Kernel 4: WMMA GEMM.  r[m][n] = x[m][n] + b[n] + sum_k h[m][k]*W[n][k]
//   Grid: (M/128, D/128). Block: 256 threads = 8 waves.
//   Wave w: (wm = w&3, wn = w>>2) owns a 32x64 subtile = 2x4 WMMA tiles.
//   Double-buffered LDS; A tile staged via CDNA5 async global->LDS copies,
//   B tile (f32->f16 convert) overlapped with WMMA compute.
// ---------------------------------------------------------------------------
#define LDSTRIDE 40   // 32 halfs + 8 pad -> 80B row pitch, conflict-free b128
#define NKSTEP   32   // D / 32 k-iterations

__global__ __launch_bounds__(256) void gemm_wmma_k(
    const _Float16* __restrict__ h, const float* __restrict__ W,
    const float* __restrict__ bproj, const float* __restrict__ x,
    float* __restrict__ r) {
  using namespace cfg;
  __shared__ __align__(16) _Float16 As[2][128 * LDSTRIDE];
  __shared__ __align__(16) _Float16 Bs[2][128 * LDSTRIDE];

  const int tid  = threadIdx.x;
  const int lane = tid & 31;
  const int lo   = lane & 15;          // col / row-within-16
  const int hi   = lane >> 4;          // 0: lanes 0-15, 1: lanes 16-31
  const int wid  = tid >> 5;
  const int wm   = wid & 3;            // 4 waves along M
  const int wn   = wid >> 2;           // 2 waves along N
  const int bm   = blockIdx.x * 128;
  const int bn   = blockIdx.y * 128;

  // staging indices (constant per thread)
  const int a_r0 = tid >> 2;                 // A: row, 2 chunks/thread
  const int a_k0 = (tid & 3) << 3;           // A: half offset 0/8/16/24
  const int b_n0 = tid >> 3;                 // B: row (n), 4 chunks/thread
  const int b_k0 = (tid & 7) << 2;           // B: half offset 0..28

  // fire async copies of the f16 A-tile straight into LDS (no VGPR round trip)
  auto stageA = [&](int k0, int buf) {
    #pragma unroll
    for (int j = 0; j < 2; ++j) {
      int rr = a_r0 + j * 64;
      unsigned ldsa = lds_off32(&As[buf][rr * LDSTRIDE + a_k0]);
      unsigned long long ga =
          (unsigned long long)(uintptr_t)(h + (size_t)(bm + rr) * D + k0 + a_k0);
      asm volatile("global_load_async_to_lds_b128 %0, %1, off"
                   :: "v"(ldsa), "v"(ga) : "memory");
    }
  };

  v8f acc[2][4];
  #pragma unroll
  for (int i = 0; i < 2; ++i)
    #pragma unroll
    for (int j = 0; j < 4; ++j) acc[i][j] = (v8f)(0.0f);

  union V16 { v16h v; v8h h[2]; };

  // ---- prologue: stage tile 0 ----
  stageA(0, 0);
  {
    #pragma unroll
    for (int j = 0; j < 4; ++j) {
      int nn = b_n0 + j * 32;
      const float4 wv = *(const float4*)(W + (size_t)(bn + nn) * D + b_k0);
      v4h hv; hv.x = (_Float16)wv.x; hv.y = (_Float16)wv.y;
              hv.z = (_Float16)wv.z; hv.w = (_Float16)wv.w;
      *(v4h*)&Bs[0][nn * LDSTRIDE + b_k0] = hv;
    }
  }
  asm volatile("s_wait_asynccnt 0x0" ::: "memory");
  __syncthreads();

  for (int it = 0; it < NKSTEP; ++it) {
    const int cur = it & 1;
    const int nxt = cur ^ 1;
    const int k1  = (it + 1) * 32;
    const bool more = (it + 1) < NKSTEP;

    // 1) fire next tile's async A copies + issue next B global loads
    float4 wv[4];
    if (more) {
      stageA(k1, nxt);
      #pragma unroll
      for (int j = 0; j < 4; ++j)
        wv[j] = *(const float4*)(W + (size_t)(bn + b_n0 + j * 32) * D + k1 + b_k0);
    }

    // 2) WMMA on current tile (overlaps with in-flight copies/loads)
    V16 afr[2];
    #pragma unroll
    for (int tm = 0; tm < 2; ++tm) {
      int rr = wm * 32 + tm * 16 + lo;
      int kb = hi ? 8 : 0;             // lanes>=16 hold K 8-15 / 24-31
      afr[tm].h[0] = *(const v8h*)&As[cur][rr * LDSTRIDE + kb];
      afr[tm].h[1] = *(const v8h*)&As[cur][rr * LDSTRIDE + kb + 16];
    }
    V16 bfr[4];
    #pragma unroll
    for (int tn = 0; tn < 4; ++tn) {
      int nn = wn * 64 + tn * 16 + lo;
      int kb = hi ? 8 : 0;             // lanes>=16 hold K 8-15 / 24-31
      bfr[tn].h[0] = *(const v8h*)&Bs[cur][nn * LDSTRIDE + kb];
      bfr[tn].h[1] = *(const v8h*)&Bs[cur][nn * LDSTRIDE + kb + 16];
    }
    #pragma unroll
    for (int tm = 0; tm < 2; ++tm)
      #pragma unroll
      for (int tn = 0; tn < 4; ++tn)
        acc[tm][tn] = __builtin_amdgcn_wmma_f32_16x16x32_f16(
            false, afr[tm].v, false, bfr[tn].v,
            (short)0, acc[tm][tn], false, false);

    // 3) convert + store next B tile, then close the pipeline stage
    if (more) {
      #pragma unroll
      for (int j = 0; j < 4; ++j) {
        int nn = b_n0 + j * 32;
        v4h hv; hv.x = (_Float16)wv[j].x; hv.y = (_Float16)wv[j].y;
                hv.z = (_Float16)wv[j].z; hv.w = (_Float16)wv[j].w;
        *(v4h*)&Bs[nxt][nn * LDSTRIDE + b_k0] = hv;
      }
      asm volatile("s_wait_asynccnt 0x0" ::: "memory");
    }
    __syncthreads();
  }

  // --- epilogue: + b_proj + residual x, write r (C/D layout: M=v+8*hi, N=lo)
  #pragma unroll
  for (int tm = 0; tm < 2; ++tm) {
    #pragma unroll
    for (int tn = 0; tn < 4; ++tn) {
      int n = bn + wn * 64 + tn * 16 + lo;
      float bn_v = bproj[n];
      #pragma unroll
      for (int v = 0; v < 8; ++v) {
        int m = bm + wm * 32 + tm * 16 + v + hi * 8;
        size_t o = (size_t)m * D + n;
        r[o] = acc[tm][tn][v] + bn_v + x[o];
      }
    }
  }
}

// ---------------------------------------------------------------------------
// Kernel 5: in-place LayerNorm per row of d_out (4096 rows x 1024)
// ---------------------------------------------------------------------------
__global__ __launch_bounds__(256) void layernorm_k(
    float* __restrict__ r, const float* __restrict__ gamma,
    const float* __restrict__ beta) {
  using namespace cfg;
  __shared__ float ssum[256];
  __shared__ float ssq[256];
  const int row = blockIdx.x;
  const int tid = threadIdx.x;
  float4 v = *(const float4*)(r + (size_t)row * D + tid * 4);
  float s  = v.x + v.y + v.z + v.w;
  float sq = v.x * v.x + v.y * v.y + v.z * v.z + v.w * v.w;
  ssum[tid] = s; ssq[tid] = sq;
  __syncthreads();
  #pragma unroll
  for (int off = 128; off > 0; off >>= 1) {
    if (tid < off) { ssum[tid] += ssum[tid + off]; ssq[tid] += ssq[tid + off]; }
    __syncthreads();
  }
  const float inv = 1.0f / (float)D;
  const float mu  = ssum[0] * inv;
  const float var = fmaxf(ssq[0] * inv - mu * mu, 0.0f);
  const float rs  = rsqrtf(var + 1e-5f);
  float4 o;
  int n = tid * 4;
  o.x = (v.x - mu) * rs * gamma[n + 0] + beta[n + 0];
  o.y = (v.y - mu) * rs * gamma[n + 1] + beta[n + 1];
  o.z = (v.z - mu) * rs * gamma[n + 2] + beta[n + 2];
  o.w = (v.w - mu) * rs * gamma[n + 3] + beta[n + 3];
  *(float4*)(r + (size_t)row * D + n) = o;
}

// ---------------------------------------------------------------------------
extern "C" void kernel_launch(void* const* d_in, const int* in_sizes, int n_in,
                              void* d_out, int out_size, void* d_ws, size_t ws_size,
                              hipStream_t stream) {
  using namespace cfg;
  const float* x       = (const float*)d_in[0];
  const float* A_param = (const float*)d_in[1];
  const float* B_vec   = (const float*)d_in[2];
  const float* C_vec   = (const float*)d_in[3];
  const float* D_vec   = (const float*)d_in[4];
  const float* W_proj  = (const float*)d_in[5];
  const float* b_proj  = (const float*)d_in[6];
  const float* gamma   = (const float*)d_in[7];
  const float* beta    = (const float*)d_in[8];
  float* out = (float*)d_out;

  _Float16* h_ws = (_Float16*)((char*)d_ws + H_OFF);
  float* part    = (float*)((char*)d_ws + PART_OFF);
  float* pfx     = (float*)((char*)d_ws + PFX_OFF);

  // 1) chunk partials: B*NCH*D = 65536 threads
  scan_partial_k<<<(Bsz * NCH * D) / 256, 256, 0, stream>>>(x, A_param, part);
  // 2) chunk prefix: B*D = 2048 threads
  scan_prefix_k<<<(Bsz * D) / 256, 256, 0, stream>>>(part, pfx);
  // 3) finalize + gelu -> h (f16)
  scan_final_k<<<(Bsz * NCH * D) / 256, 256, 0, stream>>>(
      x, A_param, B_vec, C_vec, D_vec, pfx, h_ws);
  // 4) WMMA GEMM + bias + residual -> d_out
  gemm_wmma_k<<<dim3(M / 128, D / 128), 256, 0, stream>>>(
      h_ws, W_proj, b_proj, x, out);
  // 5) in-place LayerNorm
  layernorm_k<<<M, 256, 0, stream>>>(out, gamma, beta);
}
